// waveforms_lstm_cl_v0_8504035246641
// MI455X (gfx1250) — matrix operand order repaired and checked by reading, not compiled, not run on hardware
//
#include <hip/hip_runtime.h>
#include <math.h>

#define BB 131072
#define TT 40
#define HH 40
#define CC 4

typedef __attribute__((ext_vector_type(16))) _Float16 v16h;
typedef __attribute__((ext_vector_type(2)))  _Float16 f16x2;
typedef __attribute__((ext_vector_type(8)))  float    v8f;
typedef __attribute__((ext_vector_type(2)))  float    f2v;

#define NT   12                     // 12 N-tiles of 16 (4 gates padded 40->48)
#define SROW 52                     // LDS h-tile row stride (dwords), bank-friendly
#define PWDW (16*SROW + 16*TT)      // per-wave LDS dwords: h(832) + x(640) = 1472
#define WAVES 8

__device__ __forceinline__ f16x2 cvt2(float a, float b) {
  // v_cvt_pk_rtz_f16_f32; bit_cast __fp16x2 -> _Float16x2 (same 16-bit IEEE half)
  return __builtin_bit_cast(f16x2, __builtin_amdgcn_cvt_pkrtz(a, b));
}
__device__ __forceinline__ float fast_exp2(float a) {
#if __has_builtin(__builtin_amdgcn_exp2f)
  return __builtin_amdgcn_exp2f(a);
#else
  return exp2f(a);
#endif
}
__device__ __forceinline__ float fast_rcp(float a) {
#if __has_builtin(__builtin_amdgcn_rcpf)
  return __builtin_amdgcn_rcpf(a);
#else
  return 1.0f / a;
#endif
}
__device__ __forceinline__ float sigm(float a) {
  return fast_rcp(1.0f + fast_exp2(-1.44269504f * a));
}
__device__ __forceinline__ float tanh_f(float a) {
#if __has_builtin(__builtin_amdgcn_tanhf)
  return __builtin_amdgcn_tanhf(a);
#else
  return 2.0f * fast_rcp(1.0f + fast_exp2(-2.88539008f * a)) - 1.0f;
#endif
}

__global__ __launch_bounds__(256)
void lstm_wmma_kernel(const float* __restrict__ x,
                      const float* __restrict__ W_ih,
                      const float* __restrict__ W_hh,
                      const float* __restrict__ b_ih,
                      const float* __restrict__ b_hh,
                      const float* __restrict__ W_fc,
                      const float* __restrict__ b_fc,
                      float* __restrict__ out)
{
  __shared__ float lds[WAVES * PWDW];  // 47,104 B; first 6400 dwords alias W_hh stage

  const int tid  = threadIdx.x;
  const int lane = tid & 31;
  const int wv   = tid >> 5;
  const int rh   = (lane >> 4) & 1;    // which 16-lane half
  const int ln   = lane & 15;          // column/row index within a 16-tile
  const int b0   = (blockIdx.x * WAVES + wv) * 16;

  // ---- stage W_hh (160x40 f32) into LDS, fully coalesced ----
  #pragma unroll
  for (int j = 0; j < 25; ++j)
    lds[tid + 256 * j] = W_hh[tid + 256 * j];
  __syncthreads();

  // ---- build register-resident f16 B operands (gate-padded W_hh^T) ----
  // B 32x16 f16 layout (this code's contiguous k order, same as A): VGPR v, lanes0-15:
  // K=2v,2v+1 (lo,hi), N=lane; lanes16-31: K=16+2v,17+2v, N=lane-16.
  v16h  Wv[NT][2];
  float bvec[NT], wvec[NT];
  #pragma unroll
  for (int nt = 0; nt < NT; ++nt) {
    const int  q  = nt / 3;
    const int  lc = (nt % 3) * 16 + ln;     // column within padded gate (0..47)
    const bool vn = (lc < HH);
    const int  n  = q * HH + (vn ? lc : 0); // row of W_hh (gate unit)
    bvec[nt] = vn ? (b_ih[n] + b_hh[n]) : 0.0f;
    wvec[nt] = vn ? W_ih[n] : 0.0f;
    #pragma unroll
    for (int ks = 0; ks < 2; ++ks) {
      #pragma unroll
      for (int v = 0; v < 8; ++v) {
        const int kb = ks * 32 + rh * 16 + 2 * v;
        float w0 = (vn && (kb     < HH)) ? lds[n * HH + kb    ] : 0.0f;
        float w1 = (vn && (kb + 1 < HH)) ? lds[n * HH + kb + 1] : 0.0f;
        f16x2 p = cvt2(w0, w1);
        Wv[nt][ks][2 * v    ] = p.x;
        Wv[nt][ks][2 * v + 1] = p.y;
      }
    }
  }
  __syncthreads();  // all waves done reading the W stage region (aliased below)

  // ---- per-wave LDS: zero h tile, preload x tile (coalesced) ----
  const int pwb   = wv * PWDW;
  const int xbase = pwb + 16 * SROW;
  #pragma unroll
  for (int j = 0; j < 26; ++j)
    lds[pwb + lane + 32 * j] = 0.0f;                                 // h0 = 0
  #pragma unroll
  for (int j = 0; j < 20; ++j)
    lds[xbase + lane + 32 * j] = x[(size_t)b0 * TT + lane + 32 * j]; // 16 rows of x

  v8f cc[3];
  #pragma unroll
  for (int ct = 0; ct < 3; ++ct)
    #pragma unroll
    for (int r = 0; r < 8; ++r) cc[ct][r] = 0.0f;

  // ---- time recurrence ----
  #pragma unroll 1
  for (int t = 0; t < TT; ++t) {
    __syncthreads();

    // C init: bias + x_t * w_in  (per accumulator row r -> batch row r+8*rh)
    float xa[8];
    #pragma unroll
    for (int r = 0; r < 8; ++r)
      xa[r] = lds[xbase + (r + 8 * rh) * TT + t];

    v8f acc[NT];
    #pragma unroll
    for (int nt = 0; nt < NT; ++nt)
      #pragma unroll
      for (int r = 0; r < 8; ++r)
        acc[nt][r] = bvec[nt] + xa[r] * wvec[nt];

    // A operands: h (f32 in LDS) -> packed f16, 16-bit A 16x32 layout
    v16h A0, A1;
    #pragma unroll
    for (int v = 0; v < 8; ++v) {
      const int kp = 2 * v + 16 * rh;  // K pair base, kstep0
      f2v hv = *reinterpret_cast<const f2v*>(&lds[pwb + ln * SROW + kp]);
      f16x2 p0 = cvt2(hv.x, hv.y);
      A0[2 * v] = p0.x;  A0[2 * v + 1] = p0.y;

      const int  kq = kp + 32;                              // kstep1 (pad K>=40 -> 0)
      const bool vk = (kq < HH);
      f2v hw = *reinterpret_cast<const f2v*>(&lds[pwb + ln * SROW + (vk ? kq : 0)]);
      f16x2 p1 = cvt2(vk ? hw.x : 0.0f, vk ? hw.y : 0.0f);
      A1[2 * v] = p1.x;  A1[2 * v + 1] = p1.y;
    }

    // gates += h @ W_hh^T   (24 x v_wmma_f32_16x16x32_f16)
    #pragma unroll
    for (int nt = 0; nt < NT; ++nt) {
      acc[nt] = __builtin_amdgcn_wmma_f32_16x16x32_f16(
          false, A0, false, Wv[nt][0], (short)0, acc[nt], false, false);
      acc[nt] = __builtin_amdgcn_wmma_f32_16x16x32_f16(
          false, A1, false, Wv[nt][1], (short)0, acc[nt], false, false);
    }

    // activations + state update; store new h back to LDS (f32)
    #pragma unroll
    for (int ct = 0; ct < 3; ++ct) {
      #pragma unroll
      for (int r = 0; r < 8; ++r) {
        float ig = sigm(acc[ct    ][r]);
        float fg = sigm(acc[3 + ct][r]);
        float gg = tanh_f(acc[6 + ct][r]);
        float og = sigm(acc[9 + ct][r]);
        float cn = fg * cc[ct][r] + ig * gg;
        cc[ct][r] = cn;
        float hn = og * tanh_f(cn);
        lds[pwb + (r + 8 * rh) * SROW + ct * 16 + ln] = hn;
      }
    }
  }

  // ---- epilogue: logits = h_T @ W_fc^T + b_fc, plus argmax ----
  __syncthreads();
  if (lane < 16) {
    const int gb = b0 + lane;
    float best = -3.402823466e38f;
    int   bi   = 0;
    #pragma unroll
    for (int c4 = 0; c4 < CC; ++c4) {
      float s = b_fc[c4];
      #pragma unroll
      for (int k = 0; k < HH; ++k)
        s += lds[pwb + lane * SROW + k] * W_fc[c4 * HH + k];
      out[(size_t)gb * CC + c4] = s;
      if (s > best) { best = s; bi = c4; }
    }
    out[(size_t)BB * CC + gb] = (float)bi;  // argmax, stored in output buffer
  }
}

extern "C" void kernel_launch(void* const* d_in, const int* in_sizes, int n_in,
                              void* d_out, int out_size, void* d_ws, size_t ws_size,
                              hipStream_t stream) {
  (void)in_sizes; (void)n_in; (void)out_size; (void)d_ws; (void)ws_size;
  const float* x    = (const float*)d_in[0];
  const float* W_ih = (const float*)d_in[1];
  const float* W_hh = (const float*)d_in[2];
  const float* b_ih = (const float*)d_in[3];
  const float* b_hh = (const float*)d_in[4];
  const float* W_fc = (const float*)d_in[5];
  const float* b_fc = (const float*)d_in[6];

  dim3 grid(BB / (16 * WAVES));   // 1024 blocks x 256 threads (8 waves, 16 rows each)
  lstm_wmma_kernel<<<grid, 256, 0, stream>>>(x, W_ih, W_hh, b_ih, b_hh, W_fc, b_fc,
                                             (float*)d_out);
}
